// MultiHeadMambaBlock_79130477461819
// MI455X (gfx1250) — matrix operand order repaired and checked by reading, not compiled, not run on hardware
//
#include <hip/hip_runtime.h>
#include <hip/hip_bf16.h>
#include <math.h>

typedef __attribute__((ext_vector_type(16))) _Float16     v16h;
typedef __attribute__((ext_vector_type(8)))  float        v8f;
typedef __attribute__((ext_vector_type(4)))  unsigned int v4u;
typedef __attribute__((ext_vector_type(8)))  int          v8i;
typedef __attribute__((ext_vector_type(4)))  int          v4i;

#define DIM_   512
#define DINNER 1024
#define BATCH  4
#define SEQL   2048
#define MROWS  (BATCH * SEQL)   // 8192
#define DSTATE 16
#define DTRANK 32

// Tensor Data Mover availability (device pass only; host pass lacks the builtin)
#if defined(__has_builtin)
#  if __has_builtin(__builtin_amdgcn_tensor_load_to_lds)
#    define HAVE_TDM 1
#  else
#    define HAVE_TDM 0
#  endif
#else
#  define HAVE_TDM 0
#endif
// therock-10.0 headers (amdgpu-toolchain) imply the 6-arg builtin form
#if __has_include(<hip/amd_detail/amd_gfx1250_TDM.h>)
#  define TDM_6ARG 1
#else
#  define TDM_6ARG 0
#endif

// ---------------------------------------------------------------------------
// f32 -> f16 convert (weights)
// ---------------------------------------------------------------------------
__global__ __launch_bounds__(256) void f32_to_f16_kernel(const float* __restrict__ in,
                                                         _Float16* __restrict__ out, int n) {
    int i = blockIdx.x * 256 + threadIdx.x;
    if (i < n) out[i] = (_Float16)in[i];
}

// ---------------------------------------------------------------------------
// LayerNorm over last dim (n), one block per row, 256 threads (8 wave32).
// Optional residual add. Output fp32 or fp16.
// ---------------------------------------------------------------------------
template <typename OutT>
__global__ __launch_bounds__(256) void ln_kernel(const float* __restrict__ x,
                                                 const float* __restrict__ add, // nullable
                                                 const float* __restrict__ w,
                                                 const float* __restrict__ b,
                                                 OutT* __restrict__ out, int n) {
    const int row  = blockIdx.x;
    const int tid  = threadIdx.x;
    const int lane = tid & 31;
    const int wv   = tid >> 5;
    const float* xr = x + (size_t)row * n;
    const float* ar = add ? add + (size_t)row * n : nullptr;

    float s = 0.f, ss = 0.f;
    for (int i = tid; i < n; i += 256) {
        float v = xr[i] + (ar ? ar[i] : 0.f);
        s += v; ss += v * v;
    }
    for (int o = 16; o > 0; o >>= 1) {
        s  += __shfl_xor(s, o, 32);
        ss += __shfl_xor(ss, o, 32);
    }
    __shared__ float rbuf[16];
    __shared__ float stats[2];
    if (lane == 0) { rbuf[wv] = s; rbuf[8 + wv] = ss; }
    __syncthreads();
    if (tid == 0) {
        float S = 0.f, SS = 0.f;
        for (int i = 0; i < 8; ++i) { S += rbuf[i]; SS += rbuf[8 + i]; }
        float mu  = S / n;
        float var = SS / n - mu * mu;
        stats[0] = mu;
        stats[1] = rsqrtf(var + 1e-5f);
    }
    __syncthreads();
    const float mu = stats[0], rstd = stats[1];
    for (int i = tid; i < n; i += 256) {
        float v = xr[i] + (ar ? ar[i] : 0.f);
        out[(size_t)row * n + i] = (OutT)((v - mu) * rstd * w[i] + b[i]);
    }
}

// ---------------------------------------------------------------------------
// TDM: DMA a 2-D tile (tile_rows x tile_k f16 elements, row stride = row_stride
// elements) from global memory into LDS at lds_off. Issued by one wave,
// tracked with TENSORcnt.  D# layout per CDNA5 ISA ch.8 (CDNA5 ISA).
// ---------------------------------------------------------------------------
#if HAVE_TDM
__device__ inline void tdm_load_2d(unsigned lds_off, const void* gptr,
                                   unsigned tile_k, unsigned tile_rows,
                                   unsigned long long row_stride) {
    unsigned long long ga = (unsigned long long)(uintptr_t)gptr;
    v4u g0;
    g0[0] = 1u;                                          // count=1, user descriptor
    g0[1] = lds_off;                                     // lds_addr [63:32]
    g0[2] = (unsigned)(ga & 0xFFFFFFFFu);                // global_addr lo
    g0[3] = (unsigned)((ga >> 32) & 0x1FFFFFFu)          // global_addr hi (57-bit)
          | (2u << 30);                                  // type = 2 ("image")
    const unsigned td0 = tile_k;                         // tensor_dim0 (tile always in-bounds)
    const unsigned td1 = tile_rows;                      // tensor_dim1
    v8i g1;
    g1[0] = (int)(1u << 16);                             // data_size=1 -> 2 bytes; no mask/pad
    g1[1] = (int)((td0 & 0xFFFFu) << 16);                // [63:48] = tensor_dim0[15:0]
    g1[2] = (int)((td0 >> 16) | ((td1 & 0xFFFFu) << 16));
    g1[3] = (int)((td1 >> 16) | (tile_k << 16));         // [127:112] tile_dim0
    g1[4] = (int)(tile_rows & 0xFFFFu);                  // [143:128] tile_dim1, tile_dim2=0
    g1[5] = (int)(row_stride & 0xFFFFFFFFull);           // [191:160] stride0 lo
    g1[6] = (int)((row_stride >> 32) & 0xFFFFull);       // [207:192] stride0 hi, stride1 lo16=0
    g1[7] = 0;
    v4i z4 = {0, 0, 0, 0};
#if TDM_6ARG
    v8i z8 = {0, 0, 0, 0, 0, 0, 0, 0};
    __builtin_amdgcn_tensor_load_to_lds(g0, g1, z4, z4, z8, 0);
#else
    __builtin_amdgcn_tensor_load_to_lds(g0, g1, z4, z4, 0);
#endif
}
#endif

// ---------------------------------------------------------------------------
// WMMA GEMM: C[M,N] = A[M,K](f16, row stride lda) * W[N,K]^T (f16) + bias
// Block tile 128x64, 256 threads = 8 waves (4 M x 2 N), wave tile 32x32:
// 4x v_wmma_f32_16x16x32_f16 per K-step with A/B fragment reuse.
// Staging: TDM double-buffer (tile k+1 DMA overlaps WMMA on tile k),
// or cooperative global->LDS loads when TDM is unavailable.
// Requires M%128==0, N%64==0, K%32==0, lda%8==0 (all hold here).
// ---------------------------------------------------------------------------
__device__ inline v16h frag_splitK(const _Float16* p) {  // A layout: e0-7 -> K+0..7, e8-15 -> K+16..23
    v16h f;
    ((float4*)&f)[0] = *(const float4*)(p);
    ((float4*)&f)[1] = *(const float4*)(p + 16);
    return f;
}
__device__ inline v16h frag_contigK(const _Float16* p) { // unused: 16 contiguous K values
    v16h f;
    ((float4*)&f)[0] = *(const float4*)(p);
    ((float4*)&f)[1] = *(const float4*)(p + 8);
    return f;
}

template <typename OutT>
__global__ __launch_bounds__(256) void gemm_wmma_kernel(const _Float16* __restrict__ A, int lda,
                                                        const _Float16* __restrict__ W,
                                                        const float* __restrict__ bias, // nullable
                                                        OutT* __restrict__ C,
                                                        int M, int N, int K) {
    __shared__ __align__(16) _Float16 lA[2][128 * 32];
    __shared__ __align__(16) _Float16 lB[2][64 * 32];

    const int tid  = threadIdx.x;
    const int lane = tid & 31;
    const int wv   = tid >> 5;
    const int wm   = wv & 3;   // M sub-tile (32 rows each)
    const int wn   = wv >> 2;  // N sub-tile (32 cols each)
    const int m0   = blockIdx.x * 128;
    const int n0   = blockIdx.y * 64;

    const int r16 = lane & 15;
    const int kbA = (lane & 16) ? 8  : 0;   // A split-K base
    const int kbB = (lane & 16) ? 8  : 0;   // B split-K base

    v8f acc00 = {}, acc01 = {}, acc10 = {}, acc11 = {};

    auto do_mma = [&](int buf) {
        const v16h af0 = frag_splitK (&lA[buf][(wm * 32      + r16) * 32 + kbA]);
        const v16h af1 = frag_splitK (&lA[buf][(wm * 32 + 16 + r16) * 32 + kbA]);
        const v16h bf0 = frag_splitK (&lB[buf][(wn * 32      + r16) * 32 + kbB]);
        const v16h bf1 = frag_splitK (&lB[buf][(wn * 32 + 16 + r16) * 32 + kbB]);
        acc00 = __builtin_amdgcn_wmma_f32_16x16x32_f16(false, af0, false, bf0, (short)0, acc00, false, false);
        acc01 = __builtin_amdgcn_wmma_f32_16x16x32_f16(false, af0, false, bf1, (short)0, acc01, false, false);
        acc10 = __builtin_amdgcn_wmma_f32_16x16x32_f16(false, af1, false, bf0, (short)0, acc10, false, false);
        acc11 = __builtin_amdgcn_wmma_f32_16x16x32_f16(false, af1, false, bf1, (short)0, acc11, false, false);
    };

#if HAVE_TDM
    auto issue = [&](int buf, int k0) {
        tdm_load_2d((unsigned)(uintptr_t)&lA[buf][0],
                    &A[(size_t)m0 * lda + k0], 32u, 128u, (unsigned long long)lda);
        tdm_load_2d((unsigned)(uintptr_t)&lB[buf][0],
                    &W[(size_t)n0 * K + k0], 32u, 64u, (unsigned long long)K);
    };
    if (wv == 0) issue(0, 0);
    int i = 0;
    for (int k0 = 0; k0 < K; k0 += 32, ++i) {
        const int cur = i & 1;
        if (wv == 0) __builtin_amdgcn_s_wait_tensorcnt(0);   // DMA into cur complete
        __syncthreads();                                     // cur visible to all waves
        if (k0 + 32 < K && wv == 0) issue(cur ^ 1, k0 + 32); // overlap next DMA with math
        do_mma(cur);
        __syncthreads();                                     // done reading cur before reuse
    }
#else
    // cooperative fallback: A tile 4096 halves (16/thread), B tile 2048 halves (8/thread)
    const int arow = tid >> 1, acol = (tid & 1) * 16;
    const int brow = tid >> 2, bcol = (tid & 3) * 8;
    for (int k0 = 0; k0 < K; k0 += 32) {
        *(float4*)&lA[0][arow * 32 + acol]     = *(const float4*)&A[(size_t)(m0 + arow) * lda + k0 + acol];
        *(float4*)&lA[0][arow * 32 + acol + 8] = *(const float4*)&A[(size_t)(m0 + arow) * lda + k0 + acol + 8];
        *(float4*)&lB[0][brow * 32 + bcol]     = *(const float4*)&W[(size_t)(n0 + brow) * K   + k0 + bcol];
        __syncthreads();
        do_mma(0);
        __syncthreads();
    }
#endif

    // D layout: VGPR r -> row r (lanes 0-15) / row r+8 (lanes 16-31); col = lane%16
    const int mb0 = m0 + wm * 32 + ((lane & 16) ? 8 : 0);
    const int mb1 = mb0 + 16;
    const int nc0 = n0 + wn * 32 + r16;
    const int nc1 = nc0 + 16;
    const float bv0 = bias ? bias[nc0] : 0.f;
    const float bv1 = bias ? bias[nc1] : 0.f;
#pragma unroll
    for (int r = 0; r < 8; ++r) {
        C[(size_t)(mb0 + r) * N + nc0] = (OutT)(acc00[r] + bv0);
        C[(size_t)(mb0 + r) * N + nc1] = (OutT)(acc01[r] + bv1);
        C[(size_t)(mb1 + r) * N + nc0] = (OutT)(acc10[r] + bv0);
        C[(size_t)(mb1 + r) * N + nc1] = (OutT)(acc11[r] + bv1);
    }
}

// ---------------------------------------------------------------------------
// Causal depthwise conv (width 4) + SiLU on xm = xz[..., :1024]
// ---------------------------------------------------------------------------
__global__ __launch_bounds__(256) void conv_silu_kernel(const _Float16* __restrict__ xz, // (M, 2048)
                                                        const float* __restrict__ cw,   // (1024,1,4)
                                                        const float* __restrict__ cb,   // (1024)
                                                        _Float16* __restrict__ xm,      // (M, 1024)
                                                        int L) {
    int idx = blockIdx.x * 256 + threadIdx.x;
    if (idx >= MROWS * DINNER) return;
    const int d  = idx & (DINNER - 1);
    const int ml = idx >> 10;         // b*L + l
    const int l  = ml & (SEQL - 1);
    const int b  = ml >> 11;
    float acc = cb[d];
#pragma unroll
    for (int j = 0; j < 4; ++j) {
        int lj = l - 3 + j;
        if (lj >= 0)
            acc += cw[d * 4 + j] * (float)xz[((size_t)(b * L + lj)) * (2 * DINNER) + d];
    }
    float s = acc / (1.f + expf(-acc));   // SiLU
    xm[(size_t)ml * DINNER + d] = (_Float16)s;
}

// ---------------------------------------------------------------------------
// Selective scan. One state n per lane, 16 lanes per (b,d) chain, 2 chains
// per wave32, 8 waves per block -> 16 chains/block, 256 blocks for 4096 chains.
// Fuses softplus(delta), +u*D, and SiLU(z) gating; writes gated y as f16.
// ---------------------------------------------------------------------------
__global__ __launch_bounds__(256) void scan_kernel(const _Float16* __restrict__ xm,    // u     (M,1024)
                                                   const _Float16* __restrict__ dpre,  // delta (M,1024) pre-softplus
                                                   const _Float16* __restrict__ xdbl,  // (M,64): [32,48)=B, [48,64)=C
                                                   const _Float16* __restrict__ xz,    // (M,2048), z at col 1024+d
                                                   const float* __restrict__ A_log,    // (1024,16)
                                                   const float* __restrict__ Dv,       // (1024)
                                                   _Float16* __restrict__ y,           // (M,1024)
                                                   int L) {
    const int lane = threadIdx.x & 31;
    const int wv   = threadIdx.x >> 5;
    const int g    = lane >> 4;       // chain within wave
    const int n    = lane & 15;       // state index
    const int pair = blockIdx.x * 16 + wv * 2 + g;
    const int b    = pair >> 10;
    const int d    = pair & (DINNER - 1);

    const float Aa = -expf(A_log[d * DSTATE + n]);
    const float Dd = Dv[d];
    float h = 0.f;
    const size_t base = (size_t)b * L;

    for (int t = 0; t < L; ++t) {
        const size_t row = base + t;
        float dl = (float)dpre[row * DINNER + d];
        dl = (dl > 20.f) ? dl : log1pf(expf(dl));            // softplus
        const float u  = (float)xm[row * DINNER + d];
        const float Bt = (float)xdbl[row * 64 + DTRANK + n];
        const float Ct = (float)xdbl[row * 64 + DTRANK + DSTATE + n];
        const float dA = expf(dl * Aa);
        h = dA * h + (dl * u) * Bt;
        float yn = h * Ct;
        yn += __shfl_xor(yn, 1, 32);
        yn += __shfl_xor(yn, 2, 32);
        yn += __shfl_xor(yn, 4, 32);
        yn += __shfl_xor(yn, 8, 32);
        if (n == 0) {
            float yv = yn + u * Dd;
            float z  = (float)xz[row * (2 * DINNER) + DINNER + d];
            float sz = z / (1.f + expf(-z));                 // SiLU(z)
            y[row * DINNER + d] = (_Float16)(yv * sz);
        }
    }
}

// ---------------------------------------------------------------------------
// Host launcher
// ---------------------------------------------------------------------------
extern "C" void kernel_launch(void* const* d_in, const int* in_sizes, int n_in,
                              void* d_out, int out_size, void* d_ws, size_t ws_size,
                              hipStream_t stream) {
    const float* x         = (const float*)d_in[0];
    const float* norm_w    = (const float*)d_in[1];
    const float* norm_b    = (const float*)d_in[2];
    const float* in_proj_w = (const float*)d_in[3];
    const float* conv_w    = (const float*)d_in[4];
    const float* conv_b    = (const float*)d_in[5];
    const float* x_proj_w  = (const float*)d_in[6];
    const float* dt_proj_w = (const float*)d_in[7];
    const float* dt_proj_b = (const float*)d_in[8];
    const float* A_log     = (const float*)d_in[9];
    const float* Dv        = (const float*)d_in[10];
    const float* mix_out_w = (const float*)d_in[11];
    const float* match_w   = (const float*)d_in[12];
    const float* match_b   = (const float*)d_in[13];
    const float* geom_w    = (const float*)d_in[14];
    const float* geom_b    = (const float*)d_in[15];
    const float* normm_w   = (const float*)d_in[16];
    const float* normm_b   = (const float*)d_in[17];
    const float* normg_w   = (const float*)d_in[18];
    const float* normg_b   = (const float*)d_in[19];

    const int M = MROWS;           // 8192
    const int L = SEQL;

    // workspace carve-out (~125 MB, 256B aligned slabs)
    char* ws = (char*)d_ws;
    size_t off = 0;
    auto alloc = [&](size_t bytes) -> char* {
        char* p = ws + off;
        off += (bytes + 255) & ~(size_t)255;
        return p;
    };
    _Float16* h16      = (_Float16*)alloc((size_t)M * DIM_ * 2);
    _Float16* w_inp    = (_Float16*)alloc((size_t)(2 * DINNER) * DIM_ * 2);
    _Float16* w_xp     = (_Float16*)alloc((size_t)64 * DINNER * 2);
    _Float16* w_dt     = (_Float16*)alloc((size_t)DINNER * DTRANK * 2);
    _Float16* w_mix    = (_Float16*)alloc((size_t)DIM_ * DINNER * 2);
    _Float16* w_match  = (_Float16*)alloc((size_t)DIM_ * DIM_ * 2);
    _Float16* w_geom   = (_Float16*)alloc((size_t)256 * DIM_ * 2);
    _Float16* xz16     = (_Float16*)alloc((size_t)M * 2 * DINNER * 2);
    _Float16* xm16     = (_Float16*)alloc((size_t)M * DINNER * 2);
    _Float16* xdbl16   = (_Float16*)alloc((size_t)M * 64 * 2);
    _Float16* dpre16   = (_Float16*)alloc((size_t)M * DINNER * 2);
    _Float16* y16      = (_Float16*)alloc((size_t)M * DINNER * 2);
    _Float16* mo16     = (_Float16*)alloc((size_t)M * DIM_ * 2);
    float*    mm_match = (float*)   alloc((size_t)M * DIM_ * 4);
    float*    mm_geom  = (float*)   alloc((size_t)M * 256 * 4);

    auto cvt = [&](const float* src, _Float16* dst, int n) {
        f32_to_f16_kernel<<<(n + 255) / 256, 256, 0, stream>>>(src, dst, n);
    };
    cvt(in_proj_w, w_inp,   2 * DINNER * DIM_);
    cvt(x_proj_w,  w_xp,    64 * DINNER);
    cvt(dt_proj_w, w_dt,    DINNER * DTRANK);
    cvt(mix_out_w, w_mix,   DIM_ * DINNER);
    cvt(match_w,   w_match, DIM_ * DIM_);
    cvt(geom_w,    w_geom,  256 * DIM_);

    // 1) h = LN(x) -> f16
    ln_kernel<_Float16><<<M, 256, 0, stream>>>(x, nullptr, norm_w, norm_b, h16, DIM_);

    // 2) xz = h @ in_proj_w^T   (8192 x 2048, K=512)
    gemm_wmma_kernel<_Float16><<<dim3(M / 128, (2 * DINNER) / 64), 256, 0, stream>>>(
        h16, DIM_, w_inp, nullptr, xz16, M, 2 * DINNER, DIM_);

    // 3) xm = SiLU(causal_dwconv(xz[:, :1024]))
    conv_silu_kernel<<<(M * DINNER + 255) / 256, 256, 0, stream>>>(xz16, conv_w, conv_b, xm16, L);

    // 4) x_dbl = xm @ x_proj_w^T   (8192 x 64, K=1024)
    gemm_wmma_kernel<_Float16><<<dim3(M / 128, 64 / 64), 256, 0, stream>>>(
        xm16, DINNER, w_xp, nullptr, xdbl16, M, 64, DINNER);

    // 5) delta_pre = dt @ dt_proj_w^T + dt_proj_b   (8192 x 1024, K=32; dt = x_dbl[:, :32], lda=64)
    gemm_wmma_kernel<_Float16><<<dim3(M / 128, DINNER / 64), 256, 0, stream>>>(
        xdbl16, 64, w_dt, dt_proj_b, dpre16, M, DINNER, DTRANK);

    // 6) selective scan (+softplus, +u*D, *SiLU(z)) -> y16
    scan_kernel<<<(BATCH * DINNER) / 16, 256, 0, stream>>>(
        xm16, dpre16, xdbl16, xz16, A_log, Dv, y16, L);

    // 7) mamba_out = y @ mix_out_w^T   (8192 x 512, K=1024)
    gemm_wmma_kernel<_Float16><<<dim3(M / 128, DIM_ / 64), 256, 0, stream>>>(
        y16, DINNER, w_mix, nullptr, mo16, M, DIM_, DINNER);

    // 8) mm_match = mamba_out @ match_w^T + match_b   (fp32)
    gemm_wmma_kernel<float><<<dim3(M / 128, DIM_ / 64), 256, 0, stream>>>(
        mo16, DIM_, w_match, match_b, mm_match, M, DIM_, DIM_);

    // 9) mm_geom = mamba_out @ geom_w^T + geom_b   (fp32)
    gemm_wmma_kernel<float><<<dim3(M / 128, 256 / 64), 256, 0, stream>>>(
        mo16, DIM_, w_geom, geom_b, mm_geom, M, 256, DIM_);

    // 10) y_match = LN(x + mm_match), y_geom = LN(mm_geom)  -> d_out (fp32, concatenated)
    float* out_match = (float*)d_out;
    float* out_geom  = (float*)d_out + (size_t)M * DIM_;
    ln_kernel<float><<<M, 256, 0, stream>>>(mm_match, x, normm_w, normm_b, out_match, DIM_);
    ln_kernel<float><<<M, 256, 0, stream>>>(mm_geom, nullptr, normg_w, normg_b, out_geom, 256);
}
